// AAttnQuant_34411277975940
// MI455X (gfx1250) — matrix operand order repaired and checked by reading, not compiled, not run on hardware
//
#include <hip/hip_runtime.h>

// ---------------------------------------------------------------------------
// Area-attention block for MI455X (gfx1250, wave32, WMMA + TDM).
// Fragment-native global layouts (all WMMA operands are contiguous-per-lane
// b128 loads); weight tiles staged to LDS by the Tensor Data Mover; softmax
// row-sums computed by WMMA against a ones-fragment; row-max via ds_swizzle.
// ---------------------------------------------------------------------------

typedef __bf16 bf16;
typedef __attribute__((ext_vector_type(16))) __bf16 v16bf;
typedef __attribute__((ext_vector_type(8)))  __bf16 v8bf;
typedef __attribute__((ext_vector_type(8)))  float  v8f;
typedef __attribute__((ext_vector_type(4)))  unsigned int u32x4;
typedef __attribute__((ext_vector_type(4)))  int i32x4v;
typedef __attribute__((ext_vector_type(8)))  int i32x8v;

static __device__ inline v8f wmma_bf16f32(v16bf a, v16bf b, v8f c) {
  // D = A(16x32 bf16) * B(32x16 bf16) + C(16x16 f32)
  return __builtin_amdgcn_wmma_f32_16x16x32_bf16(
      false, a, false, b, (short)0, c, false, false);
}

// XOR-reduce max across the 16-lane row groups of a C/D fragment using
// ds_swizzle_b32 (group-of-32 pattern: and=0x1f, or=0, xor=off<<10).
static __device__ inline float rowmax16(float x) {
  float y = x, o;
  o = __builtin_bit_cast(float, __builtin_amdgcn_ds_swizzle(__builtin_bit_cast(int, y), 0x041F));
  y = fmaxf(y, o);
  o = __builtin_bit_cast(float, __builtin_amdgcn_ds_swizzle(__builtin_bit_cast(int, y), 0x081F));
  y = fmaxf(y, o);
  o = __builtin_bit_cast(float, __builtin_amdgcn_ds_swizzle(__builtin_bit_cast(int, y), 0x101F));
  y = fmaxf(y, o);
  o = __builtin_bit_cast(float, __builtin_amdgcn_ds_swizzle(__builtin_bit_cast(int, y), 0x201F));
  y = fmaxf(y, o);
  return y;
}

// A fragment (16x32 bf16): element (m,k) at base[m*strideM + k*strideK].
// ISA layout: lanes 0-15 rows, K 0-7 & 16-23; lanes 16-31 same rows, K 8-15 & 24-31.
static __device__ inline v16bf load_a_frag(const bf16* base, long strideM, long strideK) {
  const int lane = threadIdx.x & 31;
  const int row  = lane & 15;
  const int kb   = (lane < 16) ? 0 : 8;
  v16bf f;
#pragma unroll
  for (int e = 0; e < 16; ++e) {
    int j = e >> 1, sub = e & 1;
    int k = (j < 4) ? (kb + 2 * j + sub) : (16 + kb + 2 * (j - 4) + sub);
    f[e] = base[(long)row * strideM + (long)k * strideK];
  }
  return f;
}

// B fragment (32x16 bf16): element (k,n) at base[k*strideK + n*strideN].
// ISA layout: lanes 0-15 -> K 0-7 & 16-23, lanes 16-31 -> K 8-15 & 24-31; N = lane&15.
static __device__ inline v16bf load_b_frag(const bf16* base, long strideK, long strideN) {
  const int lane = threadIdx.x & 31;
  const int col  = lane & 15;
  const int koff = (lane < 16) ? 0 : 8;
  v16bf f;
#pragma unroll
  for (int e = 0; e < 16; ++e)
    f[e] = base[(long)((e < 8) ? (koff + e) : (16 + koff + e - 8)) * strideK + (long)col * strideN];
  return f;
}

// ---------------------------------------------------------------------------
// GEMM: out[M,4096] = A[M,256](bf16 row-major) * Bt[4096,256](bf16 n-major)
//       + bias.  Block = 128 threads (4 waves); wave tile = 64x32.
// A tile (64x256, 32KB) staged to LDS once per block via the Tensor Data
// Mover; B straight from global/L2.  Compile-time epilogue variants.
// ---------------------------------------------------------------------------
template <bool WF, bool WBF, bool BFT>
__global__ __launch_bounds__(128) void gemm_bias_wmma(
    const bf16* __restrict__ A, const bf16* __restrict__ Bt,
    const float* __restrict__ bias, float* __restrict__ outF,
    bf16* __restrict__ outBF, int M, int Nn) {
  constexpr int K = 256;
  const int lane = threadIdx.x & 31;
  const int wv   = threadIdx.x >> 5;
  const int m0   = blockIdx.y * 64;
  const int n0   = blockIdx.x * 128 + wv * 32;
  const size_t boff = (size_t)blockIdx.z * K * Nn;
  const size_t ooff = (size_t)blockIdx.z * M * Nn;

  __shared__ bf16 alds[64 * 256];   // 32 KB weight tile

#if __has_builtin(__builtin_amdgcn_tensor_load_to_lds)
  if (threadIdx.x < 32) {           // wave 0 issues one TDM 2D descriptor
    unsigned long long ga = (unsigned long long)(const void*)(A + (size_t)m0 * K);
    unsigned ldsoff = (unsigned)(size_t)(void*)alds;  // LDS byte offset
    u32x4 g0;                       // count=1, lds_addr, global_addr, type=2
    g0[0] = 1u;
    g0[1] = ldsoff;
    g0[2] = (unsigned)(ga & 0xFFFFFFFFull);
    g0[3] = ((unsigned)(ga >> 32) & 0x01FFFFFFu) | (2u << 30);
    i32x8v g1;                      // data_size=2B, tensor 256x64, tile 256x64
    g1[0] = (1 << 16);
    g1[1] = (int)(256u << 16);      // tensor_dim0 = 256
    g1[2] = (int)(64u << 16);       // tensor_dim1 = 64
    g1[3] = (int)(256u << 16);      // tile_dim0 = 256
    g1[4] = 64;                     // tile_dim1 = 64
    g1[5] = 256;                    // tensor_dim0_stride = 256
    g1[6] = 0;
    g1[7] = 0;
    i32x4v gz = {0, 0, 0, 0};
#if __clang_major__ >= 23
    i32x8v gz8 = {0, 0, 0, 0, 0, 0, 0, 0};
    __builtin_amdgcn_tensor_load_to_lds(g0, g1, gz, gz, gz8, 0);
#else
    __builtin_amdgcn_tensor_load_to_lds(g0, g1, gz, gz, 0);
#endif
    __builtin_amdgcn_s_wait_tensorcnt(0);
  }
#else
  {                                 // fallback: cooperative vector copy
    const uint4* src = (const uint4*)(A + (size_t)m0 * K);
    uint4* dst = (uint4*)alds;
    for (int i = threadIdx.x; i < 64 * 256 / 8; i += 128) dst[i] = src[i];
  }
#endif
  __syncthreads();

  v8f acc[4][2] = {};

  for (int k0 = 0; k0 < K; k0 += 32) {
    __builtin_prefetch(Bt + boff + (size_t)n0 * K + k0 + 32, 0, 1);
    v16bf a[4], b2[2];
#pragma unroll
    for (int i = 0; i < 4; ++i)
      a[i] = load_a_frag(alds + (size_t)(i * 16) * K + k0, K, 1);
#pragma unroll
    for (int j2 = 0; j2 < 2; ++j2)
      b2[j2] = load_b_frag(Bt + boff + (size_t)(n0 + j2 * 16) * K + k0, 1, K);
#pragma unroll
    for (int i = 0; i < 4; ++i)
#pragma unroll
      for (int j2 = 0; j2 < 2; ++j2)
        acc[i][j2] = wmma_bf16f32(a[i], b2[j2], acc[i][j2]);
  }

#pragma unroll
  for (int i = 0; i < 4; ++i)
#pragma unroll
    for (int j2 = 0; j2 < 2; ++j2) {
      const int mb = m0 + i * 16 + 8 * (lane >> 4);
      const int n  = n0 + j2 * 16 + (lane & 15);
      if constexpr (BFT) {          // packed 16B store, contiguous in m
        v8bf pk;
#pragma unroll
        for (int j = 0; j < 8; ++j) pk[j] = (bf16)(acc[i][j2][j] + bias[mb + j]);
        *(v8bf*)(outBF + ooff + (size_t)n * M + mb) = pk;
      } else {
#pragma unroll
        for (int j = 0; j < 8; ++j) {
          float val = acc[i][j2][j] + bias[mb + j];
          if constexpr (WF)  outF[ooff + (size_t)(mb + j) * Nn + n] = val;
          if constexpr (WBF) outBF[ooff + (size_t)(mb + j) * Nn + n] = (bf16)val;
        }
      }
    }
}

// ---------------------------------------------------------------------------
// Depthwise 5x5 conv, pad 2, on v4 (f32 [B*C, 64,64]).  Bandwidth bound.
// ---------------------------------------------------------------------------
__global__ __launch_bounds__(256) void dwconv5x5(
    const float* __restrict__ v4, const float* __restrict__ Wpe,
    const float* __restrict__ bpe, float* __restrict__ pp) {
  const int plane = blockIdx.x;            // b*256 + c
  const int c = plane & 255;
  const float* src = v4 + (size_t)plane * 4096;
  float* dst = pp + (size_t)plane * 4096;
  float wloc[25];
#pragma unroll
  for (int i = 0; i < 25; ++i) wloc[i] = Wpe[c * 25 + i];
  const float bb = bpe[c];
  for (int px = threadIdx.x; px < 4096; px += 256) {
    int h = px >> 6, w = px & 63;
    float s = bb;
#pragma unroll
    for (int dy = -2; dy <= 2; ++dy) {
      int hh = h + dy;
      if (hh < 0 || hh > 63) continue;
#pragma unroll
      for (int dx = -2; dx <= 2; ++dx) {
        int ww = w + dx;
        if (ww < 0 || ww > 63) continue;
        s += wloc[(dy + 2) * 5 + (dx + 2)] * src[hh * 64 + ww];
      }
    }
    dst[px] = s;
  }
}

// ---------------------------------------------------------------------------
// Flash attention over one (batch, area, head): Na=1024 keys, hd=32.
// Block = 4 waves; each wave owns 16 q-rows, iterates 64-key tiles.
//   S = q*k^T (4 WMMAs), row-max via ds_swizzle XOR-reduce, P -> LDS ->
//   A-fragments, O += P*V (4 WMMAs), row-sum += P*ones (2 WMMAs).
// qkT: n-major bf16 [B, 4096, 512]; v: channel-major bf16 [B, 256, 4096];
// out: channel-major f32.
// ---------------------------------------------------------------------------
__global__ __launch_bounds__(128) void area_attn_wmma(
    const bf16* __restrict__ qkT, const bf16* __restrict__ v,
    float* __restrict__ out) {
  const int lane = threadIdx.x & 31;
  const int wv   = threadIdx.x >> 5;
  const int qt   = blockIdx.x;          // 0..15   (q tile of 64 rows)
  const int bah  = blockIdx.y;          // 0..127  (ba*HEADS + head)
  const int head = bah & 7;
  const int ba   = bah >> 3;
  const int b    = ba >> 2;
  const int area = ba & 3;
  const int N = 4096, NA = 1024, CH2 = 512;

  const bf16* qbase = qkT + ((size_t)b * N + area * NA) * CH2 + head * 32;
  const bf16* kbase = qbase + 256;
  const bf16* vbase = v   + ((size_t)b * 256 + head * 32) * N + area * NA;
  float*      obase = out + ((size_t)b * 256 + head * 32) * N + area * NA;

  const int qr = qt * 64 + wv * 16;     // this wave's q-row base within area

  __shared__ bf16 plds[4][16][64];      // per-wave P staging tile (2KB each)
  bf16* myp = &plds[wv][0][0];

  // q fragment: element (m=row, k=d) at qbase[(qr+row)*512 + d]
  v16bf qa = load_a_frag(qbase + (size_t)qr * CH2, CH2, 1);

  v16bf onesb;                          // all-ones B fragment for row sums
#pragma unroll
  for (int e = 0; e < 16; ++e) onesb[e] = (bf16)1.0f;

  v8f o0 = {}, o1 = {}, rs = {};
  float rmax[8];
#pragma unroll
  for (int j = 0; j < 8; ++j) rmax[j] = -3.0e38f;

  const float scale = 0.17677669529663687f;   // 1/sqrt(32)

  for (int kt = 0; kt < 16; ++kt) {
    const int k0 = kt * 64;
    __builtin_prefetch(kbase + (size_t)(k0 + 64) * CH2, 0, 1);
    __builtin_prefetch(vbase + k0 + 64, 0, 1);

    // S tile 16x64: 4 fragments of q * k^T
    v8f s[4];
#pragma unroll
    for (int t = 0; t < 4; ++t) {
      v16bf kb = load_b_frag(kbase + (size_t)(k0 + 16 * t) * CH2, 1, CH2);
      v8f zero = {};
      s[t] = wmma_bf16f32(qa, kb, zero);
    }

    // online softmax: row max over 64 keys, rescale accumulators
    float fac[8];
#pragma unroll
    for (int j = 0; j < 8; ++j) {
      float a0 = s[0][j] * scale, a1 = s[1][j] * scale;
      float a2 = s[2][j] * scale, a3 = s[3][j] * scale;
      float mx = rowmax16(fmaxf(fmaxf(a0, a1), fmaxf(a2, a3)));
      float nm = fmaxf(rmax[j], mx);
      fac[j] = __expf(rmax[j] - nm);
      rmax[j] = nm;
      o0[j] *= fac[j];
      o1[j] *= fac[j];
      rs[j] *= fac[j];
      s[0][j] = __expf(a0 - nm);
      s[1][j] = __expf(a1 - nm);
      s[2][j] = __expf(a2 - nm);
      s[3][j] = __expf(a3 - nm);
    }

    // Restage P (D-layout) into row-major LDS [16][64] -> A-fragments.
#pragma unroll
    for (int t = 0; t < 4; ++t)
#pragma unroll
      for (int j = 0; j < 8; ++j) {
        int row = j + 8 * (lane >> 4);
        int col = 16 * t + (lane & 15);
        myp[row * 64 + col] = (bf16)s[t][j];
      }
    asm volatile("s_wait_dscnt 0" ::: "memory");
    v16bf pa0 = load_a_frag(myp,      64, 1);   // keys k0..k0+31
    v16bf pa1 = load_a_frag(myp + 32, 64, 1);   // keys k0+32..k0+63

    // V fragments: element (k=keyrow, n=d) at vbase[d*N + k0 + keyrow]
    v16bf vb00 = load_b_frag(vbase + k0,                       1, N);
    v16bf vb01 = load_b_frag(vbase + k0 + 32,                  1, N);
    v16bf vb10 = load_b_frag(vbase + (size_t)16 * N + k0,      1, N);
    v16bf vb11 = load_b_frag(vbase + (size_t)16 * N + k0 + 32, 1, N);

    o0 = wmma_bf16f32(pa0, vb00, o0);
    o0 = wmma_bf16f32(pa1, vb01, o0);
    o1 = wmma_bf16f32(pa0, vb10, o1);
    o1 = wmma_bf16f32(pa1, vb11, o1);
    rs = wmma_bf16f32(pa0, onesb, rs);          // row sums of P
    rs = wmma_bf16f32(pa1, onesb, rs);
  }

  // epilogue: divide by row sums, packed float4 stores (rows contiguous in j)
  const int d = lane & 15;
  const int r0 = qr + 8 * (lane >> 4);
  float w0[8], w1[8];
#pragma unroll
  for (int j = 0; j < 8; ++j) {
    float inv = 1.0f / rs[j];
    w0[j] = o0[j] * inv;
    w1[j] = o1[j] * inv;
  }
  *(float4*)(obase + (size_t)d * N + r0)            = make_float4(w0[0], w0[1], w0[2], w0[3]);
  *(float4*)(obase + (size_t)d * N + r0 + 4)        = make_float4(w0[4], w0[5], w0[6], w0[7]);
  *(float4*)(obase + (size_t)(d + 16) * N + r0)     = make_float4(w1[0], w1[1], w1[2], w1[3]);
  *(float4*)(obase + (size_t)(d + 16) * N + r0 + 4) = make_float4(w1[4], w1[5], w1[6], w1[7]);
}

// ---------------------------------------------------------------------------
// Conversions / transposes (bandwidth passes)
// ---------------------------------------------------------------------------
__global__ void cvt_f32_bf16(const float* __restrict__ in,
                             bf16* __restrict__ out, long n) {
  long i = (long)blockIdx.x * blockDim.x + threadIdx.x;
  long st = (long)gridDim.x * blockDim.x;
  for (; i < n; i += st) out[i] = (bf16)in[i];
}

// [rows, cols] f32 -> [cols, rows] bf16, tiled through LDS. grid.z = batch.
__global__ __launch_bounds__(256) void transpose_cvt_bf16(
    const float* __restrict__ in, bf16* __restrict__ out, int rows, int cols) {
  __shared__ float t[32][33];
  const int r0 = blockIdx.y * 32, c0 = blockIdx.x * 32;
  const int tx = threadIdx.x, ty = threadIdx.y;   // (32, 8)
  const float* src = in + (size_t)blockIdx.z * rows * cols;
  bf16* dst = out + (size_t)blockIdx.z * rows * cols;
#pragma unroll
  for (int i = 0; i < 4; ++i)
    t[ty + 8 * i][tx] = src[(size_t)(r0 + ty + 8 * i) * cols + c0 + tx];
  __syncthreads();
#pragma unroll
  for (int i = 0; i < 4; ++i)
    dst[(size_t)(c0 + ty + 8 * i) * rows + r0 + tx] = (bf16)t[tx][ty + 8 * i];
}

// u[n,c] = bf16(a[c,n] + b[c,n]), tiled transpose.  grid.z = batch.
__global__ __launch_bounds__(256) void add_transpose_cvt_bf16(
    const float* __restrict__ a, const float* __restrict__ bsrc,
    bf16* __restrict__ out, int rows, int cols) {
  __shared__ float t[32][33];
  const int r0 = blockIdx.y * 32, c0 = blockIdx.x * 32;
  const int tx = threadIdx.x, ty = threadIdx.y;
  const size_t base = (size_t)blockIdx.z * rows * cols;
#pragma unroll
  for (int i = 0; i < 4; ++i) {
    size_t idx = base + (size_t)(r0 + ty + 8 * i) * cols + c0 + tx;
    t[ty + 8 * i][tx] = a[idx] + bsrc[idx];
  }
  __syncthreads();
#pragma unroll
  for (int i = 0; i < 4; ++i)
    out[base + (size_t)(c0 + ty + 8 * i) * rows + r0 + tx] = (bf16)t[tx][ty + 8 * i];
}

// ---------------------------------------------------------------------------
// Host-side orchestration
// ---------------------------------------------------------------------------
extern "C" void kernel_launch(void* const* d_in, const int* in_sizes, int n_in,
                              void* d_out, int out_size, void* d_ws, size_t ws_size,
                              hipStream_t stream) {
  const int B = 4, C = 256, N = 4096;     // H=W=64, AREA=4, HEADS=8, hd=32
  (void)in_sizes; (void)n_in; (void)out_size; (void)ws_size;

  const float* x     = (const float*)d_in[0];
  const float* Wqk   = (const float*)d_in[1];
  const float* bqk   = (const float*)d_in[2];
  const float* Wv    = (const float*)d_in[3];
  const float* bv    = (const float*)d_in[4];
  const float* Wpe   = (const float*)d_in[5];
  const float* bpe   = (const float*)d_in[6];
  const float* Wproj = (const float*)d_in[7];
  const float* bproj = (const float*)d_in[8];
  float* outp = (float*)d_out;

  size_t off = 0;
  auto carve = [&](size_t bytes) -> void* {
    void* p = (char*)d_ws + off;
    off += (bytes + 255) & ~(size_t)255;
    return p;
  };
  bf16*  xbT      = (bf16*)carve((size_t)B * N * C * 2);        // [B,N,C]   8 MB
  bf16*  wqk_bf   = (bf16*)carve((size_t)512 * 256 * 2);
  bf16*  wv_bf    = (bf16*)carve((size_t)256 * 256 * 2);
  bf16*  wproj_bf = (bf16*)carve((size_t)256 * 256 * 2);
  bf16*  qkT_bf   = (bf16*)carve((size_t)B * N * 512 * 2);      // [B,N,512] 16 MB
  float* v4f      = (float*)carve((size_t)B * C * N * 4);       // [B,C,N]   16 MB
  bf16*  v_bf     = (bf16*)carve((size_t)B * C * N * 2);        // [B,C,N]   8 MB
  float* ppf      = (float*)carve((size_t)B * C * N * 4);       // [B,C,N]   16 MB
  float* attnf    = (float*)carve((size_t)B * C * N * 4);       // [B,C,N]   16 MB
  bf16*  u_bfT    = (bf16*)carve((size_t)B * N * C * 2);        // [B,N,C]   8 MB

  // 1) x -> bf16, transposed to n-major [B,N,C]; weights -> bf16 row-major
  transpose_cvt_bf16<<<dim3(N / 32, C / 32, B), dim3(32, 8), 0, stream>>>(
      x, xbT, C, N);
  cvt_f32_bf16<<<512, 256, 0, stream>>>(Wqk, wqk_bf, 512L * 256);
  cvt_f32_bf16<<<256, 256, 0, stream>>>(Wv, wv_bf, 256L * 256);
  cvt_f32_bf16<<<256, 256, 0, stream>>>(Wproj, wproj_bf, 256L * 256);

  // 2) QK projection -> bf16 transposed [B,N,512] (fragment-native for attn)
  gemm_bias_wmma<false, true, true><<<dim3(N / 128, 512 / 64, B), 128, 0, stream>>>(
      wqk_bf, xbT, bqk, nullptr, qkT_bf, 512, N);

  // 3) V projection -> f32 [B,C,N] (dwconv) + bf16 [B,C,N] (attention V)
  gemm_bias_wmma<true, true, false><<<dim3(N / 128, 256 / 64, B), 128, 0, stream>>>(
      wv_bf, xbT, bv, v4f, v_bf, 256, N);

  // 4) positional depthwise conv
  dwconv5x5<<<B * C, 256, 0, stream>>>(v4f, Wpe, bpe, ppf);

  // 5) flash attention per (batch,area,head) x 16 q-tiles
  area_attn_wmma<<<dim3(16, 128), 128, 0, stream>>>(qkT_bf, v_bf, attnf);

  // 6) u[n,c] = bf16(attn + pp), transposed to n-major for proj GEMM
  add_transpose_cvt_bf16<<<dim3(N / 32, C / 32, B), dim3(32, 8), 0, stream>>>(
      attnf, ppf, u_bfT, C, N);

  // 7) output projection straight into d_out (channel-major == NCHW)
  gemm_bias_wmma<true, false, false><<<dim3(N / 128, 256 / 64, B), 128, 0, stream>>>(
      wproj_bf, u_bfT, bproj, outp, nullptr, 256, N);
}
